// CausalSelfAttention_48936857371297
// MI455X (gfx1250) — matrix operand order repaired and checked by reading, not compiled, not run on hardware
//
#include <hip/hip_runtime.h>
#include <hip/hip_bf16.h>

// MI455X / gfx1250 causal self-attention.
// bf16 WMMA (v_wmma_f32_16x16x32_bf16) with f32 accumulate everywhere,
// f32 softmax stats, flash-attention online softmax.
// Global->LDS staging uses GLOBAL_LOAD_ASYNC_TO_LDS_B128 (ASYNCcnt) where the
// copy is transform-free; transposes are register-packed b64/b32 LDS stores.
// All GEMM tiles are double-buffered to overlap staging with the WMMA stream.

typedef __attribute__((ext_vector_type(16))) __bf16 bf16x16;
typedef __attribute__((ext_vector_type(8)))  __bf16 bf16x8;
typedef __attribute__((ext_vector_type(4)))  __bf16 bf16x4;
typedef __attribute__((ext_vector_type(2)))  __bf16 bf16x2;
typedef __attribute__((ext_vector_type(8)))  float  f32x8;

#define WMMA_BF16(a, b, c) \
  __builtin_amdgcn_wmma_f32_16x16x32_bf16(false, (a), false, (b), (short)0, (c), false, false)

static __device__ inline bf16x16 concat8(bf16x8 lo, bf16x8 hi) {
  return __builtin_shufflevector(lo, hi, 0,1,2,3,4,5,6,7,8,9,10,11,12,13,14,15);
}

// A fragment (16x32 bf16): lane row m=lane&15; elements are two contiguous
// 8-element K-chunks at k = half*8 and k = 16+half*8 (ISA 7.12.2).
static __device__ inline bf16x16 frag_a(const __bf16* rowp, int half) {
  const __bf16* p = rowp + half * 8;
  bf16x8 lo = *(const bf16x8*)p;
  bf16x8 hi = *(const bf16x8*)(p + 16);
  return concat8(lo, hi);
}

// B fragment (32x16 bf16): lane col n=lane&15; two 8-element K-chunks at
// k = half*8 and k = 16+half*8 when the tile is stored [n][k] in LDS.
static __device__ inline bf16x16 frag_b(const __bf16* p) {
  bf16x8 lo = *(const bf16x8*)p;
  bf16x8 hi = *(const bf16x8*)(p + 16);
  return concat8(lo, hi);
}

// ---- async global->LDS copy (16B per lane), ASYNCcnt-tracked ---------------
// LDS address goes in the first (VDST) VGPR; low 32 bits of a generic LDS pointer are the
// LDS byte offset (flat aperture encoding), so truncation is the LDS address.
static __device__ inline void async_copy16(const void* lds_ptr, const void* gptr) {
  unsigned l = (unsigned)(uintptr_t)lds_ptr;
  unsigned long long g = (unsigned long long)(uintptr_t)gptr;
  asm volatile("global_load_async_to_lds_b128 %0, %1, off" :: "v"(l), "v"(g) : "memory");
}
static __device__ inline void wait_async0() {
  asm volatile("s_wait_asynccnt 0x0" ::: "memory");
}

static __device__ inline f32x8 zero8() {
  f32x8 z = {0.f, 0.f, 0.f, 0.f, 0.f, 0.f, 0.f, 0.f};
  return z;
}

// ---------------------------------------------------------------- casts ----
__global__ __launch_bounds__(256) void cast_f32_to_bf16(const float* __restrict__ s,
                                                        __bf16* __restrict__ d, int n) {
  int i = (blockIdx.x * 256 + threadIdx.x) * 4;
  if (i < n) {
    float4 v = *(const float4*)(s + i);
    bf16x4 o;
    o[0] = (__bf16)v.x; o[1] = (__bf16)v.y; o[2] = (__bf16)v.z; o[3] = (__bf16)v.w;
    *(bf16x4*)(d + i) = o;
  }
}

// ----------------------------------------------------------------- GEMM ----
#define BM 128
#define BN 128
#define BK 64
#define LDK 72              // padded k-stride (elements); 144B rows, 16B aligned
#define TILE_ELEMS (128 * LDK)

// A tile 128x64: 64B per thread = 4 async b128 chunks, no transform.
static __device__ inline void issue_A_async(const __bf16* A, int lda, int m0, int k0,
                                            __bf16* As, int tid) {
  int ar = tid >> 1;
  int ac = (tid & 1) * 32;
  const __bf16* g = A + (size_t)(m0 + ar) * lda + k0 + ac;
  __bf16* l = &As[ar * LDK + ac];
#pragma unroll
  for (int c = 0; c < 4; ++c) async_copy16(l + c * 8, g + c * 8);
}

// B tile 64(k) x 128(n): each thread owns 4n x (4k + 4k) and transposes in
// registers; LDS writes are packed b64 (4 contiguous k per n).
static __device__ inline void load_B_regs(const __bf16* B, int ldb, int n0, int k0,
                                          int tid, bf16x4 br[8]) {
  int n4 = (tid & 31) * 4;
  int kq = (tid >> 5) * 4;
#pragma unroll
  for (int h = 0; h < 2; ++h)
#pragma unroll
    for (int kk = 0; kk < 4; ++kk)
      br[h * 4 + kk] = *(const bf16x4*)(B + (size_t)(k0 + kq + h * 32 + kk) * ldb + n0 + n4);
}
static __device__ inline void store_B_lds(__bf16* Bs, int tid, const bf16x4 br[8]) {
  int n4 = (tid & 31) * 4;
  int kq = (tid >> 5) * 4;
#pragma unroll
  for (int h = 0; h < 2; ++h)
#pragma unroll
    for (int i = 0; i < 4; ++i) {
      bf16x4 w;
      w[0] = br[h * 4 + 0][i]; w[1] = br[h * 4 + 1][i];
      w[2] = br[h * 4 + 2][i]; w[3] = br[h * 4 + 3][i];
      *(bf16x4*)&Bs[(n4 + i) * LDK + kq + h * 32] = w;
    }
}

// 8 waves as 4(M) x 2(N); wave tile 32x64 = 2x4 WMMA subtiles, 16 WMMA/K-step.
static __device__ inline void gemm_block(const __bf16* A, int lda,
                                         const __bf16* B, int ldb, int K,
                                         int m0, int n0,
                                         __bf16* As, __bf16* Bs,   // [2][TILE_ELEMS]
                                         f32x8 acc[2][4]) {
  int tid = threadIdx.x, lane = tid & 31, wave = tid >> 5;
  int wm = wave & 3, wn = wave >> 2;
  int half = lane >> 4, lrow = lane & 15;

  issue_A_async(A, lda, m0, 0, As, tid);
  {
    bf16x4 br[8];
    load_B_regs(B, ldb, n0, 0, tid, br);
    store_B_lds(Bs, tid, br);
  }
  wait_async0();
  __syncthreads();

  const int NK = K / BK;
  for (int i = 0; i < NK; ++i) {
    int cur = i & 1, nxt = cur ^ 1;
    const __bf16* Asc = As + cur * TILE_ELEMS;
    const __bf16* Bsc = Bs + cur * TILE_ELEMS;
    bf16x4 br[8];
    bool more = (i + 1 < NK);
    if (more) {                       // stage next tile while WMMAs run
      issue_A_async(A, lda, m0, (i + 1) * BK, As + nxt * TILE_ELEMS, tid);
      load_B_regs(B, ldb, n0, (i + 1) * BK, tid, br);
    }
#pragma unroll
    for (int ks = 0; ks < 2; ++ks) {
      bf16x16 af[2], bfm[4];
#pragma unroll
      for (int mt = 0; mt < 2; ++mt)
        af[mt] = frag_a(&Asc[(wm * 32 + mt * 16 + lrow) * LDK + ks * 32], half);
#pragma unroll
      for (int nt = 0; nt < 4; ++nt)
        bfm[nt] = frag_b(&Bsc[(wn * 64 + nt * 16 + lrow) * LDK + ks * 32 + half * 8]);
#pragma unroll
      for (int mt = 0; mt < 2; ++mt)
#pragma unroll
        for (int nt = 0; nt < 4; ++nt)
          acc[mt][nt] = WMMA_BF16(af[mt], bfm[nt], acc[mt][nt]);
    }
    if (more) store_B_lds(Bs + nxt * TILE_ELEMS, tid, br);
    wait_async0();
    __syncthreads();
  }
}

// qkv = x @ w_qkv scattered to Q/K/V [B,H,T,D] bf16; 1/sqrt(D) folded into Q.
__global__ __launch_bounds__(256) void gemm_qkv_kernel(const __bf16* __restrict__ xb,
                                                       const __bf16* __restrict__ wb,
                                                       __bf16* __restrict__ Qo,
                                                       __bf16* __restrict__ Ko,
                                                       __bf16* __restrict__ Vo) {
  __shared__ __bf16 As[2 * TILE_ELEMS];
  __shared__ __bf16 Bs[2 * TILE_ELEMS];
  int m0 = blockIdx.x * BM;   // over B*T = 8192
  int n0 = blockIdx.y * BN;   // over 3C  = 3072
  f32x8 acc[2][4];
#pragma unroll
  for (int mt = 0; mt < 2; ++mt)
#pragma unroll
    for (int nt = 0; nt < 4; ++nt) acc[mt][nt] = zero8();
  gemm_block(xb, 1024, wb, 3072, 1024, m0, n0, As, Bs, acc);

  int tid = threadIdx.x, lane = tid & 31, wave = tid >> 5;
  int wm = wave & 3, wn = wave >> 2, half = lane >> 4, lrow = lane & 15;
#pragma unroll
  for (int mt = 0; mt < 2; ++mt)
#pragma unroll
    for (int nt = 0; nt < 4; ++nt)
#pragma unroll
      for (int r = 0; r < 8; ++r) {
        int gm = m0 + wm * 32 + mt * 16 + r + half * 8;
        int gn = n0 + wn * 64 + nt * 16 + lrow;
        float v = acc[mt][nt][r];
        int mat = gn >> 10, cc = gn & 1023;
        int h = cc >> 6, d = cc & 63;
        int bb = gm >> 11, t = gm & 2047;
        size_t idx = (((size_t)bb * 16 + h) * 2048 + t) * 64 + d;
        if (mat == 0)      Qo[idx] = (__bf16)(v * 0.125f);
        else if (mat == 1) Ko[idx] = (__bf16)v;
        else               Vo[idx] = (__bf16)v;
      }
}

// y = attn @ w_proj -> f32 output
__global__ __launch_bounds__(256) void gemm_proj_kernel(const __bf16* __restrict__ ab,
                                                        const __bf16* __restrict__ wb,
                                                        float* __restrict__ out) {
  __shared__ __bf16 As[2 * TILE_ELEMS];
  __shared__ __bf16 Bs[2 * TILE_ELEMS];
  int m0 = blockIdx.x * BM;
  int n0 = blockIdx.y * BN;
  f32x8 acc[2][4];
#pragma unroll
  for (int mt = 0; mt < 2; ++mt)
#pragma unroll
    for (int nt = 0; nt < 4; ++nt) acc[mt][nt] = zero8();
  gemm_block(ab, 1024, wb, 1024, 1024, m0, n0, As, Bs, acc);

  int tid = threadIdx.x, lane = tid & 31, wave = tid >> 5;
  int wm = wave & 3, wn = wave >> 2, half = lane >> 4, lrow = lane & 15;
#pragma unroll
  for (int mt = 0; mt < 2; ++mt)
#pragma unroll
    for (int nt = 0; nt < 4; ++nt)
#pragma unroll
      for (int r = 0; r < 8; ++r) {
        int gm = m0 + wm * 32 + mt * 16 + r + half * 8;
        int gn = n0 + wn * 64 + nt * 16 + lrow;
        out[(size_t)gm * 1024 + gn] = acc[mt][nt][r];
      }
}

// ------------------------------------------------------- flash attention ----
// grid: (B*H, T/128). 8 waves; wave w owns q rows [qt0+16w, +16).
// KV tiles (32 keys) double-buffered: K via async-to-LDS, V transposed via
// register-packed b32 stores.
__global__ __launch_bounds__(256) void flash_attn_kernel(const __bf16* __restrict__ Q,
                                                         const __bf16* __restrict__ K,
                                                         const __bf16* __restrict__ V,
                                                         __bf16* __restrict__ O) {
  const int T_ = 2048, D_ = 64, C_ = 1024;
  const int KSZ = 32 * 72, VSZ = 64 * 40;
  int bh = blockIdx.x;
  int qt0 = blockIdx.y * 128;
  const __bf16* Qh = Q + (size_t)bh * T_ * D_;
  const __bf16* Kh = K + (size_t)bh * T_ * D_;
  const __bf16* Vh = V + (size_t)bh * T_ * D_;
  int b = bh >> 4, h = bh & 15;

  __shared__ __bf16 Ks[2 * KSZ];       // [key][d], padded rows
  __shared__ __bf16 Vt[2 * VSZ];       // [d][key], transposed, padded rows
  __shared__ __bf16 Ps[8 * 16 * 40];   // per-wave P staging (C-layout -> A-layout)

  int tid = threadIdx.x, lane = tid & 31, wave = tid >> 5;
  int half = lane >> 4, lrow = lane & 15;
  int qrow_base = qt0 + wave * 16;

  // K-tile copy: 16B per thread, transform-free -> async
  int kr = tid >> 3, kc = (tid & 7) * 8;
  // V-tile transpose: thread owns 2 keys x 4 d, writes packed key-pairs (b32)
  int vkp = (tid & 15) * 2, vdc = (tid >> 4) * 4;

  // Q fragments for this wave's 16 rows (D=64 -> two K=32 steps)
  bf16x16 aQ[2];
  {
    const __bf16* qp = Qh + (size_t)(qrow_base + lrow) * D_;
#pragma unroll
    for (int ks = 0; ks < 2; ++ks) {
      const __bf16* p = qp + ks * 32 + half * 8;
      aQ[ks] = concat8(*(const bf16x8*)p, *(const bf16x8*)(p + 16));
    }
  }

  float mrow[8], lsum[8];
  f32x8 oA[4];
#pragma unroll
  for (int r = 0; r < 8; ++r) { mrow[r] = -INFINITY; lsum[r] = 0.f; }
#pragma unroll
  for (int dt = 0; dt < 4; ++dt) oA[dt] = zero8();

  int nblocks = (qt0 + 128) >> 5;      // causal: keys < qt0+128

  // prologue: stage kv block 0 into buffer 0
  async_copy16(&Ks[kr * 72 + kc], Kh + (size_t)kr * D_ + kc);
  {
    bf16x4 v0 = *(const bf16x4*)(Vh + (size_t)vkp * D_ + vdc);
    bf16x4 v1 = *(const bf16x4*)(Vh + (size_t)(vkp + 1) * D_ + vdc);
#pragma unroll
    for (int i = 0; i < 4; ++i) {
      bf16x2 w; w[0] = v0[i]; w[1] = v1[i];
      *(bf16x2*)&Vt[(vdc + i) * 40 + vkp] = w;
    }
  }
  wait_async0();

  for (int kb = 0; kb < nblocks; ++kb) {
    int kv0 = kb << 5;
    int cur = kb & 1, nxt = cur ^ 1;
    __syncthreads();

    bf16x4 v0, v1;
    bool more = (kb + 1 < nblocks);
    if (more) {                        // stage next kv block while computing
      int kvn = kv0 + 32;
      async_copy16(&Ks[nxt * KSZ + kr * 72 + kc], Kh + (size_t)(kvn + kr) * D_ + kc);
      v0 = *(const bf16x4*)(Vh + (size_t)(kvn + vkp) * D_ + vdc);
      v1 = *(const bf16x4*)(Vh + (size_t)(kvn + vkp + 1) * D_ + vdc);
    }

    if (kv0 <= qrow_base + 15) {       // uniform per wave
      const __bf16* Ksc = &Ks[cur * KSZ];
      const __bf16* Vtc = &Vt[cur * VSZ];
      f32x8 sc[2] = {zero8(), zero8()};
#pragma unroll
      for (int ks = 0; ks < 2; ++ks)
#pragma unroll
        for (int nt = 0; nt < 2; ++nt) {
          bf16x16 bk = frag_b(&Ksc[(nt * 16 + lrow) * 72 + ks * 32 + half * 8]);
          sc[nt] = WMMA_BF16(aQ[ks], bk, sc[nt]);
        }
      float mx[8];
#pragma unroll
      for (int r = 0; r < 8; ++r) {
        int qr = qrow_base + r + half * 8;
#pragma unroll
        for (int nt = 0; nt < 2; ++nt)
          if (kv0 + nt * 16 + lrow > qr) sc[nt][r] = -INFINITY;
        float v = fmaxf(sc[0][r], sc[1][r]);
        v = fmaxf(v, __shfl_xor(v, 1, 32));
        v = fmaxf(v, __shfl_xor(v, 2, 32));
        v = fmaxf(v, __shfl_xor(v, 4, 32));
        v = fmaxf(v, __shfl_xor(v, 8, 32));
        mx[r] = v;
      }
      float rscale[8];
#pragma unroll
      for (int r = 0; r < 8; ++r) {
        float mn = fmaxf(mrow[r], mx[r]);
        rscale[r] = __expf(mrow[r] - mn);
        mrow[r] = mn;
      }
      float rs[8];
#pragma unroll
      for (int r = 0; r < 8; ++r) rs[r] = 0.f;
#pragma unroll
      for (int nt = 0; nt < 2; ++nt)
#pragma unroll
        for (int r = 0; r < 8; ++r) {
          float pv = __expf(sc[nt][r] - mrow[r]);
          Ps[wave * 640 + (r + half * 8) * 40 + nt * 16 + lrow] = (__bf16)pv;
          rs[r] += pv;
        }
#pragma unroll
      for (int r = 0; r < 8; ++r) {
        float v = rs[r];
        v += __shfl_xor(v, 1, 32);
        v += __shfl_xor(v, 2, 32);
        v += __shfl_xor(v, 4, 32);
        v += __shfl_xor(v, 8, 32);
        lsum[r] = lsum[r] * rscale[r] + v;
      }
#pragma unroll
      for (int dt = 0; dt < 4; ++dt)
#pragma unroll
        for (int r = 0; r < 8; ++r) oA[dt][r] = oA[dt][r] * rscale[r];

      // intra-wave LDS RAW: cross-lane P staging must land before A-frag reads
      asm volatile("s_wait_dscnt 0" ::: "memory");
      bf16x16 aP = frag_a(&Ps[wave * 640 + lrow * 40], half);
#pragma unroll
      for (int dt = 0; dt < 4; ++dt) {
        bf16x16 bv = frag_b(&Vtc[(dt * 16 + lrow) * 40 + half * 8]);
        oA[dt] = WMMA_BF16(aP, bv, oA[dt]);
      }
    }

    if (more) {
#pragma unroll
      for (int i = 0; i < 4; ++i) {
        bf16x2 w; w[0] = v0[i]; w[1] = v1[i];
        *(bf16x2*)&Vt[nxt * VSZ + (vdc + i) * 40 + vkp] = w;
      }
    }
    wait_async0();
  }

  // normalize, write attn output as [B,T,C] bf16 (col = h*64 + d)
#pragma unroll
  for (int r = 0; r < 8; ++r) {
    float inv = 1.0f / lsum[r];
    int t = qrow_base + r + half * 8;
    size_t base = ((size_t)(b * 2048 + t)) * C_ + h * 64;
#pragma unroll
    for (int dt = 0; dt < 4; ++dt)
      O[base + dt * 16 + lrow] = (__bf16)(oA[dt][r] * inv);
  }
}

// ---------------------------------------------------------------- launch ----
extern "C" void kernel_launch(void* const* d_in, const int* in_sizes, int n_in,
                              void* d_out, int out_size, void* d_ws, size_t ws_size,
                              hipStream_t stream) {
  (void)in_sizes; (void)n_in; (void)out_size; (void)ws_size;
  const float* x      = (const float*)d_in[0];
  const float* w_qkv  = (const float*)d_in[1];
  const float* w_proj = (const float*)d_in[2];
  float* out = (float*)d_out;

  const size_t N_X  = 4ull * 2048 * 1024;
  const size_t N_WQ = 1024ull * 3072;
  const size_t N_WP = 1024ull * 1024;
  const size_t N_H  = 4ull * 16 * 2048 * 64;

  char* p = (char*)d_ws;
  __bf16* xb  = (__bf16*)p; p += N_X * 2;
  __bf16* wqb = (__bf16*)p; p += N_WQ * 2;
  __bf16* wpb = (__bf16*)p; p += N_WP * 2;
  __bf16* Qb  = (__bf16*)p; p += N_H * 2;
  __bf16* Kb  = (__bf16*)p; p += N_H * 2;
  __bf16* Vb  = (__bf16*)p; p += N_H * 2;
  __bf16* Ab  = (__bf16*)p; p += N_X * 2;

  cast_f32_to_bf16<<<(int)(N_X  / 1024), 256, 0, stream>>>(x,      xb,  (int)N_X);
  cast_f32_to_bf16<<<(int)(N_WQ / 1024), 256, 0, stream>>>(w_qkv,  wqb, (int)N_WQ);
  cast_f32_to_bf16<<<(int)(N_WP / 1024), 256, 0, stream>>>(w_proj, wpb, (int)N_WP);

  gemm_qkv_kernel<<<dim3(64, 24), 256, 0, stream>>>(xb, wqb, Qb, Kb, Vb);
  flash_attn_kernel<<<dim3(64, 16), 256, 0, stream>>>(Qb, Kb, Vb, Ab);
  gemm_proj_kernel<<<dim3(64, 8), 256, 0, stream>>>(Ab, wpb, out);
}
